// NSAAttention_32839319945668
// MI455X (gfx1250) — matrix operand order repaired and checked by reading, not compiled, not run on hardware
//
#include <hip/hip_runtime.h>
#include <hip/hip_bf16.h>

typedef __attribute__((ext_vector_type(16))) __bf16 v16bf;
typedef __attribute__((ext_vector_type(8)))  __bf16 v8bf;
typedef __attribute__((ext_vector_type(8)))  float  v8f;

#define B_    2
#define S_    2048
#define HID   1024
#define NH    16
#define HD    64
#define NBLK  127
#define NBLKP 128
#define NSEL  16
#define SELB  64
#define WIN   512
#define SCALE_ 0.125f

// ---------------------------------------------------------------- WMMA helpers

__device__ __forceinline__ v8f wmma_bf16(v16bf a, v16bf b, v8f c) {
  return __builtin_amdgcn_wmma_f32_16x16x32_bf16(false, a, false, b, (short)0, c,
                                                 false, false);
}

__device__ __forceinline__ v16bf cat8(v8bf lo, v8bf hi) {
  return __builtin_shufflevector(lo, hi, 0, 1, 2, 3, 4, 5, 6, 7, 8, 9, 10, 11, 12,
                                 13, 14, 15);
}

// A-matrix fragment, row-major A (M x K), tile origin (row0, k0).
// ISA 7.12.2 K-pattern per lane reduces to two contiguous 8-element chunks:
// f[e] = p[khalf + e] (e<8), f[e] = p[16 + khalf + e-8], khalf = (lane>>4)*8.
__device__ __forceinline__ v16bf load_a(const __bf16* A, int lda, int row0, int k0) {
  int lane = threadIdx.x & 31;
  const __bf16* p =
      A + (size_t)(row0 + (lane & 15)) * lda + k0 + ((lane >> 4) << 3);
  v8bf lo = *(const v8bf*)p;
  v8bf hi = *(const v8bf*)(p + 16);
  return cat8(lo, hi);
}

__device__ __forceinline__ v16bf load_a_guard(const __bf16* A, int lda, int row0,
                                              int k0, int maxrow) {
  int lane = threadIdx.x & 31;
  int row  = row0 + (lane & 15);
  if (row < maxrow) {
    const __bf16* p = A + (size_t)row * lda + k0 + ((lane >> 4) << 3);
    v8bf lo = *(const v8bf*)p;
    v8bf hi = *(const v8bf*)(p + 16);
    return cat8(lo, hi);
  }
  v16bf f;
#pragma unroll
  for (int e = 0; e < 16; ++e) f[e] = (__bf16)0.f;
  return f;
}

// B-matrix fragment from a TRANSPOSED operand: M is (N x K) row-major and
// B(k, n) = M[n][k]. Two 8-element chunks per lane, same K pattern as load_a.
__device__ __forceinline__ v16bf load_b_tr(const __bf16* M, int ldk, int k0,
                                           int col0) {
  int lane = threadIdx.x & 31;
  const __bf16* p =
      M + (size_t)(col0 + (lane & 15)) * ldk + k0 + ((lane >> 4) << 3);
  return cat8(*(const v8bf*)p, *(const v8bf*)(p + 16));
}

// ---------------------------------------------------------------- small kernels

__global__ void k_f2bf(const float* __restrict__ in, __bf16* __restrict__ out, int n) {
  int i = blockIdx.x * 256 + threadIdx.x;
  if (i < n) out[i] = (__bf16)in[i];
}

// fp32 (K x N) -> bf16 transposed (N x K)
__global__ void k_f2bf_tr(const float* __restrict__ in, __bf16* __restrict__ out,
                          int K, int N) {
  __shared__ float tile[32][33];
  int k0 = blockIdx.x * 32, n0 = blockIdx.y * 32;
  int tx = threadIdx.x & 31, ty = threadIdx.x >> 5;   // 32 x 8
  for (int i = ty; i < 32; i += 8)
    tile[i][tx] = in[(size_t)(k0 + i) * N + n0 + tx];
  __syncthreads();
  for (int i = ty; i < 32; i += 8)
    out[(size_t)(n0 + i) * K + k0 + tx] = (__bf16)tile[tx][i];
}

__global__ void k_zero(float* __restrict__ p, int n) {
  int i = blockIdx.x * 256 + threadIdx.x;
  if (i < n) p[i] = 0.f;
}

// ---------------------------------------------------------------- QKV projection
// X (B*S x HID) @ W -> q/k/v in (B, NH, S, HD); also writes V^T (B, NH, HD, S).
__global__ void __launch_bounds__(256) k_qkv_proj(
    const __bf16* __restrict__ xb, const __bf16* __restrict__ wqT,
    const __bf16* __restrict__ wkT, const __bf16* __restrict__ wvT,
    __bf16* __restrict__ qb, __bf16* __restrict__ kb, __bf16* __restrict__ vb,
    __bf16* __restrict__ vtb) {
  int gw    = blockIdx.x * 8 + (threadIdx.x >> 5);   // 49152 waves
  int which = gw / 16384;                            // 0=q 1=k 2=v
  int t     = gw - which * 16384;
  int mt = t >> 6, nt = t & 63;
  const __bf16* WT  = (which == 0) ? wqT : (which == 1) ? wkT : wvT;
  __bf16*       dst = (which == 0) ? qb : (which == 1) ? kb : vb;
  int row0 = mt << 4, col0 = nt << 4;
  int lane = threadIdx.x & 31;
  v8f acc = {};
  for (int k = 0; k < HID; k += 32) {
    if (k + 32 < HID) {
      __builtin_prefetch(xb + (size_t)(row0 + (lane & 15)) * HID + k + 32, 0, 0);
      __builtin_prefetch(WT + (size_t)(col0 + (lane & 15)) * HID + k + 32, 0, 0);
    }
    acc = wmma_bf16(load_a(xb, HID, row0, k), load_b_tr(WT, HID, k, col0), acc);
  }
  int col = col0 + (lane & 15);
  int h = col >> 6, d = col & 63;
  int rbase = row0 + ((lane >> 4) << 3);
#pragma unroll
  for (int r = 0; r < 8; ++r) {
    int row = rbase + r;                 // (b, s)
    int bb = row >> 11, s = row & 2047;
    __bf16 val = (__bf16)acc[r];
    dst[(((size_t)(bb * NH + h) * S_) + s) * HD + d] = val;
    if (which == 2)
      vtb[(((size_t)(bb * NH + h) * HD) + d) * S_ + s] = val;
  }
}

// ---------------------------------------------------------------- compression GEMM
// comp_k[b,h] (127x64) = k_blk(127x2048) @ Wck + bck; comp_v stored transposed.
__global__ void __launch_bounds__(256) k_compress(
    const __bf16* __restrict__ kb, const __bf16* __restrict__ vb,
    const __bf16* __restrict__ wckT, const __bf16* __restrict__ wcvT,
    const float* __restrict__ bck, const float* __restrict__ bcv,
    __bf16* __restrict__ ck, __bf16* __restrict__ cvt) {
  int isv = blockIdx.x & 1;
  int bh  = blockIdx.x >> 1;
  int mt  = threadIdx.x >> 5;            // 8 waves -> 8 mtiles (128 rows)
  int lane = threadIdx.x & 31;
  const __bf16* A   = (isv ? vb : kb) + (size_t)bh * S_ * HD;
  const __bf16* WT  = isv ? wcvT : wckT;
  const float*  bia = isv ? bcv : bck;
  int row0 = mt << 4;
  v8f acc[4] = {{}, {}, {}, {}};
  for (int k = 0; k < 2048; k += 32) {
    v16bf a = load_a_guard(A, 1024, row0, k, NBLK);   // lda = STRIDE*HD
#pragma unroll
    for (int nt = 0; nt < 4; ++nt)
      acc[nt] = wmma_bf16(a, load_b_tr(WT, 2048, k, nt << 4), acc[nt]);
  }
  int rbase = row0 + ((lane >> 4) << 3);
#pragma unroll
  for (int nt = 0; nt < 4; ++nt) {
    int col = (nt << 4) + (lane & 15);
    float bv = bia[col];
#pragma unroll
    for (int r = 0; r < 8; ++r) {
      int row = rbase + r;
      float v = (row < NBLK) ? (acc[nt][r] + bv) : 0.f;  // zero-pad row 127
      if (isv)
        cvt[(size_t)bh * HD * NBLKP + (size_t)col * NBLKP + row] = (__bf16)v;
      else
        ck[(size_t)bh * NBLKP * HD + (size_t)row * HD + col] = (__bf16)v;
    }
  }
}

// ---------------------------------------------------------------- gates
__global__ void k_gates(const __bf16* __restrict__ qb, const float* __restrict__ Wg,
                        const float* __restrict__ bg, float* __restrict__ gates) {
  int bh = blockIdx.x;
  int d  = threadIdx.x;      // 64 threads
  const __bf16* p = qb + (size_t)bh * S_ * HD + d;
  float s = 0.f;
  for (int i = 0; i < S_; ++i) s += (float)p[(size_t)i * HD];
  __shared__ float qm[HD];
  qm[d] = s * (1.f / S_);
  __syncthreads();
  if (d < 3) {
    float acc = bg[d];
    for (int e = 0; e < HD; ++e) acc += qm[e] * Wg[e * 3 + d];
    gates[bh * 4 + d] = 1.f / (1.f + __expf(-acc));
  }
}

// ---------------------------------------------------------------- compressed attention
__global__ void __launch_bounds__(256) k_comp_attn(
    const __bf16* __restrict__ qb, const __bf16* __restrict__ ck,
    const __bf16* __restrict__ cvt, const float* __restrict__ gates,
    float* __restrict__ oacc, float* __restrict__ bimp) {
  __shared__ __bf16 ldsP[8][16 * NBLKP];   // 32 KB, wave-private slices
  int wid  = threadIdx.x >> 5;
  int gw   = blockIdx.x * 8 + wid;         // 4096 waves = B*NH*128 qtiles
  int bh   = gw >> 7;
  int qt   = gw & 127;
  int lane = threadIdx.x & 31;
  const __bf16* Q  = qb + (size_t)bh * S_ * HD;
  const __bf16* K  = ck + (size_t)bh * NBLKP * HD;
  const __bf16* Vt = cvt + (size_t)bh * HD * NBLKP;
  int q0 = qt << 4;
  v16bf aq0 = load_a(Q, HD, q0, 0);
  v16bf aq1 = load_a(Q, HD, q0, 32);

  float s[8][8];
#pragma unroll
  for (int ct = 0; ct < 8; ++ct) {
    v8f sc = {};
    sc = wmma_bf16(aq0, load_b_tr(K, HD, 0,  ct << 4), sc);
    sc = wmma_bf16(aq1, load_b_tr(K, HD, 32, ct << 4), sc);
    bool mask = (ct == 7) && ((lane & 15) == 15);      // key 127 is padding
#pragma unroll
    for (int r = 0; r < 8; ++r) s[ct][r] = mask ? -1e30f : sc[r] * SCALE_;
  }
  float m[8], l[8];
#pragma unroll
  for (int r = 0; r < 8; ++r) {
    float mx = s[0][r];
#pragma unroll
    for (int ct = 1; ct < 8; ++ct) mx = fmaxf(mx, s[ct][r]);
#pragma unroll
    for (int off = 1; off < 16; off <<= 1) mx = fmaxf(mx, __shfl_xor(mx, off, 32));
    m[r] = mx;
  }
#pragma unroll
  for (int r = 0; r < 8; ++r) {
    float sum = 0.f;
#pragma unroll
    for (int ct = 0; ct < 8; ++ct) { s[ct][r] = __expf(s[ct][r] - m[r]); sum += s[ct][r]; }
#pragma unroll
    for (int off = 1; off < 16; off <<= 1) sum += __shfl_xor(sum, off, 32);
    l[r] = sum;
  }
  int b  = bh >> 4;
  int rb = (lane >> 4) << 3;
#pragma unroll
  for (int ct = 0; ct < 8; ++ct) {
    float cs = 0.f;
#pragma unroll
    for (int r = 0; r < 8; ++r) {
      float pr = s[ct][r] / l[r];                      // normalized probability
      cs += pr;
      ldsP[wid][(rb + r) * NBLKP + (ct << 4) + (lane & 15)] = (__bf16)pr;
    }
    cs += __shfl_xor(cs, 16, 32);                      // add other row-half
    int col = (ct << 4) + (lane & 15);
    if (lane < 16 && col < NBLK)
      atomicAdd(&bimp[b * NBLKP + col], cs * (1.f / (NH * S_)));
  }
  v8f o[4] = {{}, {}, {}, {}};
#pragma unroll
  for (int ks = 0; ks < 4; ++ks) {
    v16bf ap = load_a(&ldsP[wid][0], NBLKP, 0, ks << 5);
#pragma unroll
    for (int nt = 0; nt < 4; ++nt)
      o[nt] = wmma_bf16(ap, load_b_tr(Vt, NBLKP, ks << 5, nt << 4), o[nt]);
  }
  float g0 = gates[bh * 4 + 0];
  float* dst = oacc + (size_t)bh * S_ * HD;
#pragma unroll
  for (int nt = 0; nt < 4; ++nt) {
    int col = (nt << 4) + (lane & 15);
#pragma unroll
    for (int r = 0; r < 8; ++r)
      dst[(size_t)(q0 + rb + r) * HD + col] = g0 * o[nt][r];
  }
}

// ---------------------------------------------------------------- top-k + gathers
__global__ void k_topk(const float* __restrict__ bimp, int* __restrict__ selidx) {
  if (threadIdx.x != 0) return;
  int b = blockIdx.x;
  float v[NBLK];
  for (int i = 0; i < NBLK; ++i) v[i] = bimp[b * NBLKP + i];
  for (int j = 0; j < NSEL; ++j) {
    int best = 0; float bv = v[0];
    for (int i = 1; i < NBLK; ++i) if (v[i] > bv) { bv = v[i]; best = i; }
    v[best] = -1e30f;
    for (int t = 0; t < SELB; ++t) {
      int tok = best * 16 + t;
      if (tok > S_ - 1) tok = S_ - 1;
      selidx[b * (NSEL * SELB) + j * SELB + t] = tok;
    }
  }
}

// K gather: row-major (B,NH,1024,HD) rows copied contiguously.
__global__ void k_gather_k(const __bf16* __restrict__ kb,
                           const int* __restrict__ selidx,
                           __bf16* __restrict__ ksel) {
  int rid = blockIdx.x * 4 + (threadIdx.x >> 6);   // B*NH*1024 rows
  int d   = threadIdx.x & 63;
  int b   = rid >> 14;
  int hl  = rid & 16383;
  int h   = hl >> 10;
  int l   = hl & 1023;
  int tok = selidx[b * 1024 + l];
  ksel[(size_t)rid * HD + d] = kb[((size_t)(b * NH + h) * S_ + tok) * HD + d];
}

// V^T gather: (B,NH,HD,1024) built from V^T (B,NH,HD,S); contiguous writes.
__global__ void k_gather_vt(const __bf16* __restrict__ vtb,
                            const int* __restrict__ selidx,
                            __bf16* __restrict__ vselt) {
  int row = blockIdx.x;                 // B*NH*HD rows
  int b   = row >> 10;                  // NH*HD = 1024
  const __bf16* src = vtb + (size_t)row * S_;
  __bf16*       dst = vselt + (size_t)row * 1024;
  const int*    si  = selidx + b * 1024;
  int l0 = threadIdx.x * 4;
#pragma unroll
  for (int j = 0; j < 4; ++j) dst[l0 + j] = src[si[l0 + j]];
}

// ---------------------------------------------------------------- online-softmax attention
// Flash-style attention over nk keys starting at key0; K row-major per head,
// V transposed (HD x vt_ld) per head. Accumulates g * out into oacc.
__global__ void __launch_bounds__(256) k_attn_online(
    const __bf16* __restrict__ qb, const __bf16* __restrict__ kbase,
    const __bf16* __restrict__ vtbase, int nk, int key0, long long k_bh_stride,
    int vt_ld, const float* __restrict__ gates, int gidx,
    float* __restrict__ oacc) {
  __shared__ __bf16 ldsP[8][16 * 32];     // 8 KB, wave-private P tiles
  int wid  = threadIdx.x >> 5;
  int gw   = blockIdx.x * 8 + wid;        // 4096 waves = B*NH*128 qtiles
  int bh   = gw >> 7;
  int qt   = gw & 127;
  int lane = threadIdx.x & 31;
  const __bf16* Q  = qb + (size_t)bh * S_ * HD;
  const __bf16* K  = kbase + (size_t)bh * k_bh_stride;
  const __bf16* Vt = vtbase + (size_t)bh * HD * vt_ld;
  int q0 = qt << 4;
  v16bf aq0 = load_a(Q, HD, q0, 0);
  v16bf aq1 = load_a(Q, HD, q0, 32);
  float m[8], l[8];
  v8f o[4] = {{}, {}, {}, {}};
#pragma unroll
  for (int r = 0; r < 8; ++r) { m[r] = -1e30f; l[r] = 0.f; }
  int rb = (lane >> 4) << 3;

  for (int c0 = 0; c0 < nk; c0 += 32) {
    float s[2][8];
#pragma unroll
    for (int ct = 0; ct < 2; ++ct) {
      v8f sc = {};
      sc = wmma_bf16(aq0, load_b_tr(K, HD, 0,  key0 + c0 + (ct << 4)), sc);
      sc = wmma_bf16(aq1, load_b_tr(K, HD, 32, key0 + c0 + (ct << 4)), sc);
#pragma unroll
      for (int r = 0; r < 8; ++r) s[ct][r] = sc[r] * SCALE_;
    }
#pragma unroll
    for (int r = 0; r < 8; ++r) {
      float mx = fmaxf(s[0][r], s[1][r]);
#pragma unroll
      for (int off = 1; off < 16; off <<= 1) mx = fmaxf(mx, __shfl_xor(mx, off, 32));
      float mnew = fmaxf(m[r], mx);
      float scl  = __expf(m[r] - mnew);
      m[r] = mnew;
      float p0 = __expf(s[0][r] - mnew);
      float p1 = __expf(s[1][r] - mnew);
      s[0][r] = p0; s[1][r] = p1;
      float rs = p0 + p1;
#pragma unroll
      for (int off = 1; off < 16; off <<= 1) rs += __shfl_xor(rs, off, 32);
      l[r] = l[r] * scl + rs;
#pragma unroll
      for (int nt = 0; nt < 4; ++nt) o[nt][r] *= scl;
    }
#pragma unroll
    for (int ct = 0; ct < 2; ++ct)
#pragma unroll
      for (int r = 0; r < 8; ++r)
        ldsP[wid][(rb + r) * 32 + (ct << 4) + (lane & 15)] = (__bf16)s[ct][r];
    v16bf ap = load_a(&ldsP[wid][0], 32, 0, 0);
#pragma unroll
    for (int nt = 0; nt < 4; ++nt)
      o[nt] = wmma_bf16(ap, load_b_tr(Vt, vt_ld, key0 + c0, nt << 4), o[nt]);
  }
  float g = gates[bh * 4 + gidx];
  float* dst = oacc + (size_t)bh * S_ * HD;
#pragma unroll
  for (int nt = 0; nt < 4; ++nt) {
    int col = (nt << 4) + (lane & 15);
#pragma unroll
    for (int r = 0; r < 8; ++r) {
      size_t idx = (size_t)(q0 + rb + r) * HD + col;
      dst[idx] += g * (o[nt][r] / l[r]);
    }
  }
}

// ---------------------------------------------------------------- combine + output proj
__global__ void k_combine(const float* __restrict__ oacc, __bf16* __restrict__ att) {
  int i  = blockIdx.x * 256 + threadIdx.x;       // B*S*HID
  int c  = i & 1023;
  int bs = i >> 10;
  int s  = bs & 2047;
  int b  = bs >> 11;
  int h = c >> 6, d = c & 63;
  att[i] = (__bf16)oacc[(((size_t)(b * NH + h) * S_) + s) * HD + d];
}

__global__ void __launch_bounds__(256) k_out_proj(const __bf16* __restrict__ att,
                                                  const __bf16* __restrict__ woT,
                                                  float* __restrict__ out) {
  int gw = blockIdx.x * 8 + (threadIdx.x >> 5);  // 16384 tiles
  int mt = gw >> 6, nt = gw & 63;
  int row0 = mt << 4, col0 = nt << 4;
  int lane = threadIdx.x & 31;
  v8f acc = {};
  for (int k = 0; k < HID; k += 32) {
    if (k + 32 < HID) {
      __builtin_prefetch(att + (size_t)(row0 + (lane & 15)) * HID + k + 32, 0, 0);
      __builtin_prefetch(woT + (size_t)(col0 + (lane & 15)) * HID + k + 32, 0, 0);
    }
    acc = wmma_bf16(load_a(att, HID, row0, k), load_b_tr(woT, HID, k, col0), acc);
  }
  int col   = col0 + (lane & 15);
  int rbase = row0 + ((lane >> 4) << 3);
#pragma unroll
  for (int r = 0; r < 8; ++r)
    out[(size_t)(rbase + r) * HID + col] = acc[r];
}

// ---------------------------------------------------------------- launcher

extern "C" void kernel_launch(void* const* d_in, const int* in_sizes, int n_in,
                              void* d_out, int out_size, void* d_ws, size_t ws_size,
                              hipStream_t stream) {
  (void)in_sizes; (void)n_in; (void)out_size; (void)ws_size;
  const float* x   = (const float*)d_in[0];
  const float* Wq  = (const float*)d_in[1];
  const float* Wk  = (const float*)d_in[2];
  const float* Wv  = (const float*)d_in[3];
  const float* Wo  = (const float*)d_in[4];
  const float* Wck = (const float*)d_in[5];
  const float* bck = (const float*)d_in[6];
  const float* Wcv = (const float*)d_in[7];
  const float* bcv = (const float*)d_in[8];
  const float* Wg  = (const float*)d_in[9];
  const float* bg  = (const float*)d_in[10];

  char* p = (char*)d_ws;
  auto alloc = [&](size_t bytes) {
    char* r = p;
    p += (bytes + 255) & ~(size_t)255;
    return r;
  };
  const size_t NX  = (size_t)B_ * S_ * HID;        // 4194304
  const size_t NW  = (size_t)HID * HID;            // 1048576
  const size_t NWC = (size_t)2048 * HD;            // 131072
  const size_t NQ  = (size_t)B_ * NH * S_ * HD;    // 4194304
  const size_t NC  = (size_t)B_ * NH * NBLKP * HD; // 262144
  const size_t NKS = (size_t)B_ * NH * 1024 * HD;  // 2097152

  __bf16* xb    = (__bf16*)alloc(NX * 2);
  __bf16* wqT   = (__bf16*)alloc(NW * 2);
  __bf16* wkT   = (__bf16*)alloc(NW * 2);
  __bf16* wvT   = (__bf16*)alloc(NW * 2);
  __bf16* woT   = (__bf16*)alloc(NW * 2);
  __bf16* wckT  = (__bf16*)alloc(NWC * 2);
  __bf16* wcvT  = (__bf16*)alloc(NWC * 2);
  __bf16* qb    = (__bf16*)alloc(NQ * 2);
  __bf16* kb    = (__bf16*)alloc(NQ * 2);
  __bf16* vb    = (__bf16*)alloc(NQ * 2);
  __bf16* vtb   = (__bf16*)alloc(NQ * 2);
  __bf16* ckb   = (__bf16*)alloc(NC * 2);
  __bf16* cvt   = (__bf16*)alloc(NC * 2);
  float*  oacc  = (float*)alloc(NQ * 4);
  __bf16* attb  = (__bf16*)alloc(NX * 2);
  float*  bimp  = (float*)alloc(B_ * NBLKP * 4);
  int*    seli  = (int*)alloc(B_ * 1024 * 4);
  __bf16* ksel  = (__bf16*)alloc(NKS * 2);
  __bf16* vselt = (__bf16*)alloc(NKS * 2);
  float*  gate  = (float*)alloc(B_ * NH * 4 * 4);

  // 1) conversions: x plain, all weights transposed (contiguous WMMA B loads)
  k_f2bf<<<(int)(NX / 256), 256, 0, stream>>>(x, xb, (int)NX);
  k_f2bf_tr<<<dim3(32, 32), 256, 0, stream>>>(Wq, wqT, HID, HID);
  k_f2bf_tr<<<dim3(32, 32), 256, 0, stream>>>(Wk, wkT, HID, HID);
  k_f2bf_tr<<<dim3(32, 32), 256, 0, stream>>>(Wv, wvT, HID, HID);
  k_f2bf_tr<<<dim3(32, 32), 256, 0, stream>>>(Wo, woT, HID, HID);
  k_f2bf_tr<<<dim3(64, 2), 256, 0, stream>>>(Wck, wckT, 2048, HD);
  k_f2bf_tr<<<dim3(64, 2), 256, 0, stream>>>(Wcv, wcvT, 2048, HD);
  // 2) Q/K/V projections (also emits V^T)
  k_qkv_proj<<<6144, 256, 0, stream>>>(xb, wqT, wkT, wvT, qb, kb, vb, vtb);
  // 3) compression GEMMs (comp_k row-major, comp_v transposed)
  k_compress<<<B_ * NH * 2, 256, 0, stream>>>(kb, vb, wckT, wcvT, bck, bcv, ckb, cvt);
  // 4) gates first, so attention branches accumulate gated output
  k_gates<<<B_ * NH, 64, 0, stream>>>(qb, Wg, bg, gate);
  // 5) compressed attention (+ block importance)
  k_zero<<<1, 256, 0, stream>>>(bimp, B_ * NBLKP);
  k_comp_attn<<<512, 256, 0, stream>>>(qb, ckb, cvt, gate, oacc, bimp);
  // 6) top-k block selection + K / V^T gathers
  k_topk<<<B_, 32, 0, stream>>>(bimp, seli);
  k_gather_k<<<B_ * NH * 1024 / 4, 256, 0, stream>>>(kb, seli, ksel);
  k_gather_vt<<<B_ * NH * HD, 256, 0, stream>>>(vtb, seli, vselt);
  // 7) selected attention (1024 keys) and window attention (last 512 keys)
  k_attn_online<<<512, 256, 0, stream>>>(qb, ksel, vselt, 1024, 0,
                                         (long long)1024 * HD, 1024, gate, 1, oacc);
  k_attn_online<<<512, 256, 0, stream>>>(qb, kb, vtb, WIN, S_ - WIN,
                                         (long long)S_ * HD, S_, gate, 2, oacc);
  // 8) combine into (B,S,HID) bf16, then output projection -> fp32
  k_combine<<<(int)(NX / 256), 256, 0, stream>>>(oacc, attb);
  k_out_proj<<<2048, 256, 0, stream>>>(attb, woT, (float*)d_out);
}
